// CenterLossNN_84688165142795
// MI455X (gfx1250) — matrix operand order repaired and checked by reading, not compiled, not run on hardware
//
#include <hip/hip_runtime.h>
#include <hip/hip_bf16.h>

// ---------------------------------------------------------------------------
// Problem constants (fixed by the reference)
// ---------------------------------------------------------------------------
constexpr int BATCH = 65536;
constexpr int DIN   = 512;
constexpr int NCLS  = 512;
constexpr int L0N   = 2048;
constexpr int L1N   = 1024;
constexpr int L2N   = 512;

// ---------------------------------------------------------------------------
// Vector types
// ---------------------------------------------------------------------------
typedef __bf16 v16bf  __attribute__((ext_vector_type(16)));
typedef __bf16 bf16x8 __attribute__((ext_vector_type(8)));
typedef float  v8f    __attribute__((ext_vector_type(8)));
typedef unsigned int u32x4 __attribute__((ext_vector_type(4)));
typedef int    i32x8  __attribute__((ext_vector_type(8)));
typedef int    i32x4  __attribute__((ext_vector_type(4)));

__device__ __forceinline__ v16bf concat8(bf16x8 lo, bf16x8 hi) {
    return __builtin_shufflevector(lo, hi, 0,1,2,3,4,5,6,7,8,9,10,11,12,13,14,15);
}

#if __has_builtin(__builtin_amdgcn_tensor_load_to_lds) && \
    __has_builtin(__builtin_amdgcn_s_wait_tensorcnt)
#define HAVE_TDM 1
#else
#define HAVE_TDM 0
#endif

// ---------------------------------------------------------------------------
// Workspace layout (byte offsets). x_bf16 region reused for h2 (dead after
// GEMM1; stream order guarantees safety).
// ---------------------------------------------------------------------------
constexpr size_t OFF_SUMS   = 0;                               // 512*512 f32
constexpr size_t OFF_COUNTS = (size_t)NCLS * DIN * 4;          // 512 f32
constexpr size_t OFF_SLOTS  = OFF_COUNTS + 512 * 4;            // 8 f32
constexpr int    ZERO_N     = NCLS * DIN + NCLS + 8;
constexpr size_t MiB        = (size_t)1 << 20;
constexpr size_t OFF_XBF    = 2 * MiB;     // 65536*512  bf16 (64 MiB)
constexpr size_t OFF_W0B    = 66 * MiB;    // 2048*512   bf16 ( 2 MiB)
constexpr size_t OFF_W1B    = 68 * MiB;    // 1024*2048  bf16 ( 4 MiB)
constexpr size_t OFF_W2B    = 72 * MiB;    // 512*1024   bf16 ( 1 MiB)
constexpr size_t OFF_H0B    = 80 * MiB;    // 65536*2048 bf16 (256 MiB)
constexpr size_t OFF_H1B    = 336 * MiB;   // 65536*1024 bf16 (128 MiB)
constexpr size_t OFF_H2B    = OFF_XBF;     // 65536*512  bf16 (reuse)

// slots: [0]=sum||x||^2  [1]=sum_c ||sums_c||^2/cnt  [2..4]=CE sums

// ---------------------------------------------------------------------------
// Kernel 0: zero the accumulation region
// ---------------------------------------------------------------------------
__global__ void k_zero(float* p, int n) {
    int i = blockIdx.x * blockDim.x + threadIdx.x;
    if (i < n) p[i] = 0.0f;
}

// ---------------------------------------------------------------------------
// Kernel 1: convert W0/W1/W2 to bf16
// ---------------------------------------------------------------------------
__global__ void k_cvt_weights(const float* __restrict__ W0,
                              const float* __restrict__ W1,
                              const float* __restrict__ W2,
                              __bf16* __restrict__ W0b,
                              __bf16* __restrict__ W1b,
                              __bf16* __restrict__ W2b) {
    constexpr int n0 = L0N * DIN;
    constexpr int n1 = L1N * L0N;
    constexpr int n2 = L2N * L1N;
    int i = blockIdx.x * blockDim.x + threadIdx.x;
    if (i < n0)                W0b[i]            = (__bf16)W0[i];
    else if (i < n0 + n1)      W1b[i - n0]       = (__bf16)W1[i - n0];
    else if (i < n0 + n1 + n2) W2b[i - n0 - n1]  = (__bf16)W2[i - n0 - n1];
}

// ---------------------------------------------------------------------------
// Kernel 2: scatter (per-class sums via LDS ds_add_f32), counts, sum||x||^2,
//           and x -> bf16 conversion. Grid: (16 col-slices of 32, 32 row
//           chunks of 2048). Block = 256 threads (8 row-waves x 32 cols).
// ---------------------------------------------------------------------------
__global__ __launch_bounds__(256)
void k_scatter(const float* __restrict__ x, const int* __restrict__ labels,
               float* __restrict__ sums, float* __restrict__ counts,
               float* __restrict__ slots, __bf16* __restrict__ xbf) {
    __shared__ float accLds[NCLS * 32];
    __shared__ float cntLds[NCLS];

    const int lane = threadIdx.x & 31;
    const int rw   = threadIdx.x >> 5;
    const int c0   = blockIdx.x * 32;
    const int base = blockIdx.y * 2048;

    for (int i = threadIdx.x; i < NCLS * 32; i += 256) accLds[i] = 0.0f;
    for (int i = threadIdx.x; i < NCLS;      i += 256) cntLds[i] = 0.0f;
    __syncthreads();

    float ss = 0.0f;
    for (int i = 0; i < 256; ++i) {
        const int r   = base + i * 8 + rw;
        const int lbl = labels[r];
        const float v = x[(size_t)r * DIN + c0 + lane];
        ss += v * v;
        xbf[(size_t)r * DIN + c0 + lane] = (__bf16)v;
        atomicAdd(&accLds[lbl * 32 + lane], v);          // ds_add_f32
        if (blockIdx.x == 0 && lane == 0)
            atomicAdd(&cntLds[lbl], 1.0f);
    }
    for (int off = 16; off > 0; off >>= 1) ss += __shfl_xor(ss, off, 32);
    if (lane == 0) atomicAdd(&slots[0], ss);

    __syncthreads();
    for (int idx = threadIdx.x; idx < NCLS * 32; idx += 256) {
        const float v = accLds[idx];
        if (v != 0.0f)
            atomicAdd(&sums[(size_t)(idx >> 5) * DIN + c0 + (idx & 31)], v);
    }
    if (blockIdx.x == 0)
        for (int idx = threadIdx.x; idx < NCLS; idx += 256)
            if (cntLds[idx] != 0.0f) atomicAdd(&counts[idx], cntLds[idx]);
}

// ---------------------------------------------------------------------------
// Kernel 3: center-loss second term: sum_c ||sums_c||^2 / count_c
// ---------------------------------------------------------------------------
__global__ __launch_bounds__(256)
void k_center_term(const float* __restrict__ sums,
                   const float* __restrict__ counts,
                   float* __restrict__ slots) {
    __shared__ float red[256];
    const int c = blockIdx.x;
    float p = 0.0f;
    for (int d = threadIdx.x; d < DIN; d += 256) {
        const float s = sums[(size_t)c * DIN + d];
        p += s * s;
    }
    red[threadIdx.x] = p;
    __syncthreads();
    for (int s = 128; s > 0; s >>= 1) {
        if (threadIdx.x < s) red[threadIdx.x] += red[threadIdx.x + s];
        __syncthreads();
    }
    if (threadIdx.x == 0) {
        const float cnt = counts[c];
        if (cnt > 0.0f) atomicAdd(&slots[1], red[0] / cnt);
    }
}

// ---------------------------------------------------------------------------
// TDM helpers: build a D# for a 2D bf16 tile (tileK x 128 rows) and issue
// TENSOR_LOAD_TO_LDS. Descriptor bitfields per CDNA5 ISA §8.3/§8.4:
//  group0: [1:0]=count=1, [63:32]=lds_addr, [120:64]=global_addr, type=2
//  group1: data_size=1 (2B), pad_enable, pad_interval=16 DW, pad_amount=4 DW
//          (reproduces the 40-element padded LDS row stride in hardware),
//          tensor_dim0=K, tensor_dim1=128, tile_dim0=32, tile_dim1=128,
//          tensor_dim0_stride=K.  groups 2/3 = 0 (2D tensor).
// LDS byte address = low 32 bits of the generic pointer (ISA §10.2: LDS
// aperture addresses carry the LDS offset in addr[31:0]).
// ---------------------------------------------------------------------------
#if HAVE_TDM
__device__ __forceinline__ void tdm_load_tile(const __bf16* gsrc,
                                              const void* lds_dst, int K) {
    const unsigned long long ga = (unsigned long long)(size_t)gsrc;
    const unsigned int ldsaddr  = (unsigned int)(size_t)lds_dst;

    u32x4 g0;
    g0[0] = 1u;                                           // count=1, load
    g0[1] = ldsaddr;                                      // lds_addr
    g0[2] = (unsigned int)(ga & 0xFFFFFFFFu);             // global_addr lo
    g0[3] = (unsigned int)((ga >> 32) & 0x01FFFFFFu)      // global_addr hi
          | (2u << 30);                                   // type = 2 (image)

    i32x8 g1;
    g1[0] = (int)((1u << 16)        // data_size = 2 bytes
                | (1u << 20)        // pad_enable
                | (3u << 22)        // pad_interval: 16 DWORDs (=32 bf16)
                | (3u << 25));      // pad_amount:    4 DWORDs (= 8 bf16)
    g1[1] = (int)(((unsigned)K & 0xFFFFu) << 16);                 // dim0 lo
    g1[2] = (int)((((unsigned)K >> 16) & 0xFFFFu) | (128u << 16));// dim0 hi | dim1 lo
    g1[3] = (int)(32u << 16);                                     // dim1 hi=0 | tile_dim0=32
    g1[4] = (int)128u;                                            // tile_dim1=128, tile_dim2=0
    g1[5] = (int)K;                                               // dim0_stride lo32
    g1[6] = 0;                                                    // dim0_stride hi | dim1_stride lo
    g1[7] = 0;

    i32x4 z4 = {0, 0, 0, 0};
#if defined(__clang_major__) && (__clang_major__ >= 23)
    i32x8 z8 = {0, 0, 0, 0, 0, 0, 0, 0};
    __builtin_amdgcn_tensor_load_to_lds(g0, g1, z4, z4, z8, 0);
#else
    __builtin_amdgcn_tensor_load_to_lds(g0, g1, z4, z4, 0);
#endif
}
#endif

// ---------------------------------------------------------------------------
// Kernel 4: tiled bf16 WMMA GEMM:  D(MxN, bf16) = A(MxK) * B(NxK)^T + bias
// Block tile 128x128, BK=32 (one v_wmma_f32_16x16x32_bf16 K-step).
// 8 waves (2 M x 4 N); each wave owns 4x2 v8f accumulator tiles.
// TDM path: wave 0 issues tensor_load_to_lds for the NEXT K-tile into the
// alternate LDS buffer while all waves run WMMAs on the current one;
// s_wait_tensorcnt 0 + workgroup barrier publish the buffer.
// ---------------------------------------------------------------------------
constexpr int LDSK = 40;   // padded K stride in LDS elements

__global__ __launch_bounds__(256)
void k_gemm_bf16(const __bf16* __restrict__ A, const __bf16* __restrict__ Bw,
                 const float* __restrict__ bias, __bf16* __restrict__ D,
                 int M, int N, int K) {
    __shared__ __align__(16) __bf16 sA[2][128 * LDSK];
    __shared__ __align__(16) __bf16 sB[2][128 * LDSK];

    const int tid  = threadIdx.x;
    const int lane = tid & 31;
    const int half = lane >> 4;
    const int l16  = lane & 15;
    const int wave = tid >> 5;
    const int wm   = wave >> 2;     // 0..1
    const int wn   = wave & 3;      // 0..3

    const int blockM = blockIdx.y * 128;
    const int blockN = blockIdx.x * 128;

    const __bf16* tileA = A  + (size_t)blockM * K;   // +k0 per iteration
    const __bf16* tileB = Bw + (size_t)blockN * K;

    v8f acc[4][2] = {};
    const int nk = K >> 5;

#if HAVE_TDM
    if (wave == 0) {
        tdm_load_tile(tileA, &sA[0][0], K);
        tdm_load_tile(tileB, &sB[0][0], K);
        __builtin_amdgcn_s_wait_tensorcnt(0);
    }
    __syncthreads();
#else
    const int ldRow = tid >> 1;
    const int ldSeg = (tid & 1) * 16;
#endif

    for (int i = 0; i < nk; ++i) {
        const int cur = i & 1;
#if HAVE_TDM
        if (wave == 0 && i + 1 < nk) {
            const int nxt = cur ^ 1;
            tdm_load_tile(tileA + (size_t)(i + 1) * 32, &sA[nxt][0], K);
            tdm_load_tile(tileB + (size_t)(i + 1) * 32, &sB[nxt][0], K);
        }
#else
        {   // manual staging into the current buffer
            const __bf16* srcA = tileA + (size_t)ldRow * K + i * 32 + ldSeg;
            const __bf16* srcB = tileB + (size_t)ldRow * K + i * 32 + ldSeg;
            uint4 a0 = ((const uint4*)srcA)[0];
            uint4 a1 = ((const uint4*)srcA)[1];
            uint4 b0 = ((const uint4*)srcB)[0];
            uint4 b1 = ((const uint4*)srcB)[1];
            *(uint4*)&sA[cur][ldRow * LDSK + ldSeg]     = a0;
            *(uint4*)&sA[cur][ldRow * LDSK + ldSeg + 8] = a1;
            *(uint4*)&sB[cur][ldRow * LDSK + ldSeg]     = b0;
            *(uint4*)&sB[cur][ldRow * LDSK + ldSeg + 8] = b1;
        }
        __syncthreads();
#endif
        const __bf16* sAc = &sA[cur][0];
        const __bf16* sBc = &sB[cur][0];

        // A fragments (16x32 bf16; lanes0-15: K 0..7/16..23, lanes16-31: 8..15/24..31)
        v16bf af[4];
        #pragma unroll
        for (int t = 0; t < 4; ++t) {
            const int m = wm * 64 + t * 16 + l16;
            bf16x8 lo = *(const bf16x8*)&sAc[m * LDSK + half * 8];
            bf16x8 hi = *(const bf16x8*)&sAc[m * LDSK + 16 + half * 8];
            af[t] = concat8(lo, hi);
        }
        // B fragments (32x16 bf16; lanes0-15: K 0..7/16..23, lanes16-31: 8..15/24..31)
        v16bf bfr[2];
        #pragma unroll
        for (int j = 0; j < 2; ++j) {
            const int n = wn * 32 + j * 16 + l16;
            bf16x8 lo = *(const bf16x8*)&sBc[n * LDSK + half * 8];
            bf16x8 hi = *(const bf16x8*)&sBc[n * LDSK + 16 + half * 8];
            bfr[j] = concat8(lo, hi);
        }
        #pragma unroll
        for (int t = 0; t < 4; ++t)
            #pragma unroll
            for (int j = 0; j < 2; ++j)
                acc[t][j] = __builtin_amdgcn_wmma_f32_16x16x32_bf16(
                    false, af[t], false, bfr[j], (short)0, acc[t][j],
                    false, false);

#if HAVE_TDM
        if (wave == 0 && i + 1 < nk)
            __builtin_amdgcn_s_wait_tensorcnt(0);
#endif
        __syncthreads();
    }

    // epilogue: bias add + bf16 store (C/D layout: VGPR r -> row r+8*half)
    #pragma unroll
    for (int t = 0; t < 4; ++t) {
        #pragma unroll
        for (int j = 0; j < 2; ++j) {
            const int gn = blockN + wn * 32 + j * 16 + l16;
            const float bv = bias[gn];
            const int gm0 = blockM + wm * 64 + t * 16 + half * 8;
            __bf16* dp = D + (size_t)gm0 * N + gn;
            #pragma unroll
            for (int r = 0; r < 8; ++r) {
                dp[0] = (__bf16)(acc[t][j][r] + bv);
                dp += N;
            }
        }
    }
}

// ---------------------------------------------------------------------------
// Kernel 5: cross-entropy, one wave per row, online softmax
// ---------------------------------------------------------------------------
__global__ __launch_bounds__(256)
void k_ce(const __bf16* __restrict__ h, const int* __restrict__ labels,
          float* __restrict__ slot, int N) {
    const int gw   = blockIdx.x * 8 + (threadIdx.x >> 5);
    const int lane = threadIdx.x & 31;
    const __bf16* rowp = h + (size_t)gw * N;

    float m = -3.4e38f, s = 0.0f;
    for (int j = lane; j < N; j += 32) {
        const float v  = (float)rowp[j];
        const float nm = fmaxf(m, v);
        s = s * __expf(m - nm) + __expf(v - nm);
        m = nm;
    }
    for (int off = 16; off > 0; off >>= 1) {
        const float m2 = __shfl_xor(m, off, 32);
        const float s2 = __shfl_xor(s, off, 32);
        const float nm = fmaxf(m, m2);
        s = s * __expf(m - nm) + s2 * __expf(m2 - nm);
        m = nm;
    }
    if (lane == 0) {
        const int lbl = labels[gw];
        const float nll = (m + __logf(s)) - (float)rowp[lbl];
        atomicAdd(slot, nll);
    }
}

// ---------------------------------------------------------------------------
// Kernel 6: final combine
// ---------------------------------------------------------------------------
__global__ void k_finalize(const float* __restrict__ slots,
                           const float* __restrict__ lambdas,
                           float* __restrict__ out) {
    if (threadIdx.x == 0 && blockIdx.x == 0) {
        const float center = slots[0] - slots[1];
        const float invB = 1.0f / (float)BATCH;
        out[0] = lambdas[0] * center
               + lambdas[1] * slots[2] * invB
               + lambdas[2] * slots[3] * invB
               + lambdas[3] * slots[4] * invB;
    }
}

// ---------------------------------------------------------------------------
// Host launcher
// d_in order: 0:x 1:W0 2:b0 3:W1 4:b1 5:W2 6:b2 7:lambda_values 8:labels
// ---------------------------------------------------------------------------
extern "C" void kernel_launch(void* const* d_in, const int* in_sizes, int n_in,
                              void* d_out, int out_size, void* d_ws, size_t ws_size,
                              hipStream_t stream) {
    const float* x       = (const float*)d_in[0];
    const float* W0      = (const float*)d_in[1];
    const float* b0      = (const float*)d_in[2];
    const float* W1      = (const float*)d_in[3];
    const float* b1      = (const float*)d_in[4];
    const float* W2      = (const float*)d_in[5];
    const float* b2      = (const float*)d_in[6];
    const float* lambdas = (const float*)d_in[7];
    const int*   labels  = (const int*)d_in[8];
    float* out = (float*)d_out;

    char* ws = (char*)d_ws;
    float*  sums   = (float*)(ws + OFF_SUMS);
    float*  counts = (float*)(ws + OFF_COUNTS);
    float*  slots  = (float*)(ws + OFF_SLOTS);
    __bf16* xbf    = (__bf16*)(ws + OFF_XBF);
    __bf16* W0b    = (__bf16*)(ws + OFF_W0B);
    __bf16* W1b    = (__bf16*)(ws + OFF_W1B);
    __bf16* W2b    = (__bf16*)(ws + OFF_W2B);
    __bf16* h0b    = (__bf16*)(ws + OFF_H0B);
    __bf16* h1b    = (__bf16*)(ws + OFF_H1B);
    __bf16* h2b    = (__bf16*)(ws + OFF_H2B);

    // 0) zero accumulators
    k_zero<<<(ZERO_N + 255) / 256, 256, 0, stream>>>((float*)ws, ZERO_N);

    // 1) weights -> bf16
    {
        const int total = L0N * DIN + L1N * L0N + L2N * L1N;
        k_cvt_weights<<<(total + 255) / 256, 256, 0, stream>>>(W0, W1, W2, W0b, W1b, W2b);
    }

    // 2) scatter: class sums, counts, sum||x||^2, x->bf16
    k_scatter<<<dim3(16, 32), 256, 0, stream>>>(x, labels, sums, counts, slots, xbf);

    // 3) center-loss second term
    k_center_term<<<NCLS, 256, 0, stream>>>(sums, counts, slots);

    // 4) level 0: h0 = x @ W0^T + b0   (65536 x 512 x 2048)
    k_gemm_bf16<<<dim3(L0N / 128, BATCH / 128), 256, 0, stream>>>(
        xbf, W0b, b0, h0b, BATCH, L0N, DIN);
    k_ce<<<BATCH / 8, 256, 0, stream>>>(h0b, labels, &slots[2], L0N);

    // 5) level 1: h1 = h0 @ W1^T + b1  (65536 x 2048 x 1024)
    k_gemm_bf16<<<dim3(L1N / 128, BATCH / 128), 256, 0, stream>>>(
        h0b, W1b, b1, h1b, BATCH, L1N, L0N);
    k_ce<<<BATCH / 8, 256, 0, stream>>>(h1b, labels, &slots[3], L1N);

    // 6) level 2: h2 = h1 @ W2^T + b2  (65536 x 1024 x 512)
    k_gemm_bf16<<<dim3(L2N / 128, BATCH / 128), 256, 0, stream>>>(
        h1b, W2b, b2, h2b, BATCH, L2N, L1N);
    k_ce<<<BATCH / 8, 256, 0, stream>>>(h2b, labels, &slots[4], L2N);

    // 7) combine
    k_finalize<<<1, 64, 0, stream>>>(slots, lambdas, out);
}
